// MyViT_31516470018565
// MI455X (gfx1250) — matrix operand order repaired and checked by reading, not compiled, not run on hardware
//
#include <hip/hip_runtime.h>
#include <hip/hip_bf16.h>
#include <math.h>

typedef _Float16 h8   __attribute__((ext_vector_type(8)));
typedef _Float16 v16h __attribute__((ext_vector_type(16)));
typedef float    v8f  __attribute__((ext_vector_type(8)));

// 32-bit LDS byte offset of a __shared__ object (generic -> addrspace(3) -> int).
// Two-step integer cast keeps the host pass happy (pointers are 64-bit there).
__device__ __forceinline__ unsigned lds_off(const void* p) {
    return (unsigned)(unsigned long long)(const __attribute__((address_space(3))) char*)p;
}

// CDNA5 async memory->LDS copy (16B per lane), tracked by ASYNCcnt.
__device__ __forceinline__ void async_load_b128(unsigned lds_byte_off, const void* gaddr) {
    asm volatile("global_load_async_to_lds_b128 %0, %1, off"
                 :: "v"(lds_byte_off), "v"((unsigned long long)gaddr)
                 : "memory");
}
__device__ __forceinline__ void wait_async() {
    asm volatile("s_wait_asynccnt 0x0" ::: "memory");
}

// ---------------------------------------------------------------------------
// WMMA GEMM: C[M,N] = act( alpha * A[M,K] @ B^T + bias + residual )
//   A: f16 row-major (lda), K multiple of 32.
//   B: f16 stored N x K row-major (ldb) -> contiguous-K async b128 loads.
//   Staging rows are CLAMPED to the last valid row instead of predicated:
//   out-of-range tile rows only feed discarded outputs, so loads stay
//   unconditional and branch-free; stores remain bounds-checked.
//   Batched via gridDim.z; z -> (bn = z/nH, bh = z%nH), offsets bn*s?n+bh*s?h.
//   TRANSV: epilogue stores V transposed+padded: Vt[n][h][o][224].
// Tile: 128x64 per 256-thread block (8 wave32); each wave 32x32 = 4 WMMAs/step.
// ---------------------------------------------------------------------------
template <bool F16OUT, bool F32OUT, bool GELU_ACT, bool RESID, bool TRANSV>
__global__ __launch_bounds__(256) void gemm_wmma(
    const _Float16* __restrict__ A, const _Float16* __restrict__ B,
    float* __restrict__ Cf, _Float16* __restrict__ Ch,
    const float* __restrict__ bias, const float* __restrict__ res,
    int M, int N, int K, int lda, int ldb, int ldc, float alpha,
    int nH, long sAn, long sAh, long sBn, long sBh, long sCn, long sCh)
{
    __shared__ _Float16 As[128][40];  // [m][k], +8 pad: 16B-aligned rows, bank spread
    __shared__ _Float16 Bs[64][40];   // [n][k]

    const int bz = blockIdx.z;
    const int bn = bz / nH, bh = bz % nH;
    A += (long)bn * sAn + (long)bh * sAh;
    B += (long)bn * sBn + (long)bh * sBh;
    const long coff = (long)bn * sCn + (long)bh * sCh;

    const int m0 = blockIdx.y * 128;
    const int n0 = blockIdx.x * 64;
    const int tid  = threadIdx.x;
    const int lane = tid & 31;
    const int wave = tid >> 5;
    const int waveM = (wave >> 1) << 5;   // 0,32,64,96
    const int waveN = (wave & 1) << 5;    // 0,32

    v8f acc00 = {}, acc01 = {}, acc10 = {}, acc11 = {};

    const int ar = tid >> 2;           // 0..63
    const int ak = (tid & 3) << 3;     // 0,8,16,24

    // clamped source rows (branch-free staging)
    const int gmA0 = (m0 + ar      < M) ? (m0 + ar)      : (M - 1);
    const int gmA1 = (m0 + ar + 64 < M) ? (m0 + ar + 64) : (M - 1);
    const int gnB  = (n0 + ar      < N) ? (n0 + ar)      : (N - 1);
    const _Float16* pA0 = A + (long)gmA0 * lda + ak;
    const _Float16* pA1 = A + (long)gmA1 * lda + ak;
    const _Float16* pB  = B + (long)gnB  * ldb + ak;

    const unsigned ldsA0 = lds_off(&As[ar][ak]);
    const unsigned ldsA1 = lds_off(&As[ar + 64][ak]);
    const unsigned ldsB  = lds_off(&Bs[ar][ak]);

    for (int kt = 0; kt < K; kt += 32) {
        // stage A (128x32) + B (64x32) tiles straight into LDS (ASYNCcnt path)
        async_load_b128(ldsA0, pA0 + kt);
        async_load_b128(ldsA1, pA1 + kt);
        async_load_b128(ldsB,  pB  + kt);
        wait_async();
        __syncthreads();

        // A fragments: lane<16 -> K 0..7/16..23 ; lane>=16 -> K 8..15/24..31
        const int rA  = waveM + (lane & 15);
        const int k0a = (lane < 16) ? 0 : 8;
        h8 a0lo = *(const h8*)&As[rA][k0a];
        h8 a0hi = *(const h8*)&As[rA][k0a + 16];
        v16h af0 = __builtin_shufflevector(a0lo, a0hi, 0,1,2,3,4,5,6,7,8,9,10,11,12,13,14,15);
        h8 a1lo = *(const h8*)&As[rA + 16][k0a];
        h8 a1hi = *(const h8*)&As[rA + 16][k0a + 16];
        v16h af1 = __builtin_shufflevector(a1lo, a1hi, 0,1,2,3,4,5,6,7,8,9,10,11,12,13,14,15);

        // B fragments: lane = column; lane<16 -> K 0..7/16..23 ; lane>=16 -> K 8..15/24..31
        const int k0b = (lane < 16) ? 0 : 8;
        const int c0  = waveN + (lane & 15);
        h8 b0lo = *(const h8*)&Bs[c0][k0b];
        h8 b0hi = *(const h8*)&Bs[c0][k0b + 16];
        v16h bf0 = __builtin_shufflevector(b0lo, b0hi, 0,1,2,3,4,5,6,7,8,9,10,11,12,13,14,15);
        h8 b1lo = *(const h8*)&Bs[c0 + 16][k0b];
        h8 b1hi = *(const h8*)&Bs[c0 + 16][k0b + 16];
        v16h bf1 = __builtin_shufflevector(b1lo, b1hi, 0,1,2,3,4,5,6,7,8,9,10,11,12,13,14,15);

        acc00 = __builtin_amdgcn_wmma_f32_16x16x32_f16(false, af0, false, bf0,
                                                       (short)0, acc00, false, false);
        acc01 = __builtin_amdgcn_wmma_f32_16x16x32_f16(false, af0, false, bf1,
                                                       (short)0, acc01, false, false);
        acc10 = __builtin_amdgcn_wmma_f32_16x16x32_f16(false, af1, false, bf0,
                                                       (short)0, acc10, false, false);
        acc11 = __builtin_amdgcn_wmma_f32_16x16x32_f16(false, af1, false, bf1,
                                                       (short)0, acc11, false, false);
        __syncthreads();
    }

    // Epilogue. C/D layout: VGPR i -> row i + 8*(lane>=16), col = lane&15.
    const int colW = n0 + waveN + (lane & 15);
    const int rowW = m0 + waveM + ((lane >> 4) << 3);
#pragma unroll
    for (int sm = 0; sm < 2; ++sm) {
#pragma unroll
        for (int i = 0; i < 8; ++i) {
            const int row = rowW + sm * 16 + i;
            if (row >= M) continue;
#pragma unroll
            for (int sn = 0; sn < 2; ++sn) {
                const int col = colW + sn * 16;
                if (col >= N) continue;
                float v;
                if (sm == 0) v = sn == 0 ? acc00[i] : acc01[i];
                else         v = sn == 0 ? acc10[i] : acc11[i];
                v *= alpha;
                if (bias) v += bias[col];
                if (TRANSV) {
                    // row = n*197 + s (token), col = h*64 + o
                    const long idx = (((long)(row / 197) * 12 + (col >> 6)) * 64
                                      + (col & 63)) * 224 + (row % 197);
                    Ch[idx] = (_Float16)v;
                } else {
                    const long idx = coff + (long)row * ldc + col;
                    if (RESID) v += res[idx];
                    if (GELU_ACT) v = 0.5f * v * (1.0f + erff(v * 0.70710678118f));
                    if (F32OUT) Cf[idx] = v;
                    if (F16OUT) Ch[idx] = (_Float16)v;
                }
            }
        }
    }
}

// ---------------------------------------------------------------------------
// Elementwise / prep kernels
// ---------------------------------------------------------------------------
__global__ void convert_f16_kernel(const float* __restrict__ in, _Float16* __restrict__ out,
                                   long total) {
    long idx = (long)blockIdx.x * blockDim.x + threadIdx.x;
    if (idx >= total) return;
    out[idx] = (_Float16)in[idx];
}

__global__ void im2col_kernel(const float* __restrict__ x, _Float16* __restrict__ out,
                              long total) {
    long idx = (long)blockIdx.x * blockDim.x + threadIdx.x;
    if (idx >= total) return;
    int k = (int)(idx % 768); long t = idx / 768;
    int p = (int)(t % 196);  int n = (int)(t / 196);
    int c = k >> 8, rem = k & 255, hi = rem >> 4, wi = rem & 15;
    int pi = p / 14, pj = p % 14;
    long src = (((long)n * 3 + c) * 224 + pi * 16 + hi) * 224 + pj * 16 + wi;
    out[idx] = (_Float16)x[src];
}

// Wq (L,H,64,64)[h][o][i] -> block-diagonal N x K (L,768,768)[n][k] f16
__global__ void blockdiag_kernel(const float* __restrict__ w, _Float16* __restrict__ out,
                                 long total) {
    long idx = (long)blockIdx.x * blockDim.x + threadIdx.x;
    if (idx >= total) return;
    int k = (int)(idx % 768); long t = idx / 768;
    int n = (int)(t % 768);   int l = (int)(t / 768);
    int hk = k >> 6, hn = n >> 6;
    float v = 0.0f;
    if (hk == hn) v = w[(((long)l * 12 + hn) * 64 + (n & 63)) * 64 + (k & 63)];
    out[idx] = (_Float16)v;
}

// zero the seq-pad region of Vt[n][h][o][197..223]
__global__ void zero_vt_pad_kernel(_Float16* __restrict__ vt, long total) {
    long idx = (long)blockIdx.x * blockDim.x + threadIdx.x;
    if (idx >= total) return;
    int s = 197 + (int)(idx % 27); long t = idx / 27;
    int o = (int)(t % 64); long t2 = t / 64;
    int h = (int)(t2 % 12); int n = (int)(t2 / 12);
    vt[(((long)n * 12 + h) * 64 + o) * 224 + s] = (_Float16)0.0f;
}

// OUT[n,s,d] = (s==0 ? cls[d] : TOK[n, s-1, d]) + posembed(s, d)
__global__ void assemble_tokens_kernel(const float* __restrict__ tok,
                                       const float* __restrict__ cls,
                                       float* __restrict__ out, long total) {
    long idx = (long)blockIdx.x * blockDim.x + threadIdx.x;
    if (idx >= total) return;
    int d = (int)(idx % 768); long t = idx / 768;
    int s = (int)(t % 197);   int n = (int)(t / 197);
    float fd   = (float)(d & ~1);
    float freq = powf(10000.0f, fd / 768.0f);
    float ang  = (float)s / freq;
    float pos  = (d & 1) ? cosf(ang) : sinf(ang);
    float v = (s == 0) ? cls[d] : tok[((long)n * 196 + (s - 1)) * 768 + d];
    out[idx] = v + pos;
}

// one 256-thread block per row, D = 768 fixed
__global__ void layernorm_kernel(const float* __restrict__ x, const float* __restrict__ g,
                                 const float* __restrict__ b, _Float16* __restrict__ out) {
    __shared__ float red[256];
    const int row = blockIdx.x, t = threadIdx.x;
    const float* xr = x + (long)row * 768;
    float v0 = xr[t], v1 = xr[t + 256], v2 = xr[t + 512];
    red[t] = v0 + v1 + v2;
    __syncthreads();
    for (int o = 128; o > 0; o >>= 1) { if (t < o) red[t] += red[t + o]; __syncthreads(); }
    const float mu = red[0] * (1.0f / 768.0f);
    __syncthreads();
    float d0 = v0 - mu, d1 = v1 - mu, d2 = v2 - mu;
    red[t] = d0 * d0 + d1 * d1 + d2 * d2;
    __syncthreads();
    for (int o = 128; o > 0; o >>= 1) { if (t < o) red[t] += red[t + o]; __syncthreads(); }
    const float rs = rsqrtf(red[0] * (1.0f / 768.0f) + 1e-5f);
    _Float16* orow = out + (long)row * 768;
    orow[t]       = (_Float16)(d0 * rs * g[t]       + b[t]);
    orow[t + 256] = (_Float16)(d1 * rs * g[t + 256] + b[t + 256]);
    orow[t + 512] = (_Float16)(d2 * rs * g[t + 512] + b[t + 512]);
}

// one block per score row; 197 valid cols, padded to 224 with zeros (f16 out)
__global__ void softmax_kernel(const float* __restrict__ S, _Float16* __restrict__ P) {
    __shared__ float red[256];
    const long row = blockIdx.x;
    const int t = threadIdx.x;
    const float* sr = S + row * 224;
    float v = (t < 197) ? sr[t] : -3.4e38f;
    red[t] = v;
    __syncthreads();
    for (int o = 128; o > 0; o >>= 1) { if (t < o) red[t] = fmaxf(red[t], red[t + o]); __syncthreads(); }
    const float mx = red[0];
    __syncthreads();
    float e = (t < 197) ? __expf(v - mx) : 0.0f;
    red[t] = e;
    __syncthreads();
    for (int o = 128; o > 0; o >>= 1) { if (t < o) red[t] += red[t + o]; __syncthreads(); }
    const float inv = 1.0f / red[0];
    if (t < 224) P[row * 224 + t] = (_Float16)((t < 197) ? e * inv : 0.0f);
}

__global__ void extract_cls_kernel(const float* __restrict__ out, _Float16* __restrict__ a,
                                   int total) {
    int idx = blockIdx.x * blockDim.x + threadIdx.x;
    if (idx >= total) return;
    int d = idx % 768, n = idx / 768;
    a[idx] = (_Float16)out[(long)n * 197 * 768 + d];
}

// ---------------------------------------------------------------------------
extern "C" void kernel_launch(void* const* d_in, const int* in_sizes, int n_in,
                              void* d_out, int out_size, void* d_ws, size_t ws_size,
                              hipStream_t stream) {
    (void)in_sizes; (void)n_in; (void)out_size; (void)ws_size;
    const float* x    = (const float*)d_in[0];
    const float* Wp   = (const float*)d_in[1];
    const float* bp   = (const float*)d_in[2];
    const float* cls  = (const float*)d_in[3];
    const float* ln1g = (const float*)d_in[4];
    const float* ln1b = (const float*)d_in[5];
    const float* Wq   = (const float*)d_in[6];
    const float* bq   = (const float*)d_in[7];
    const float* Wk   = (const float*)d_in[8];
    const float* bk   = (const float*)d_in[9];
    const float* Wv   = (const float*)d_in[10];
    const float* bv   = (const float*)d_in[11];
    const float* ln2g = (const float*)d_in[12];
    const float* ln2b = (const float*)d_in[13];
    const float* W1   = (const float*)d_in[14];
    const float* b1   = (const float*)d_in[15];
    const float* W2   = (const float*)d_in[16];
    const float* b2   = (const float*)d_in[17];
    const float* Wh   = (const float*)d_in[18];
    const float* bh   = (const float*)d_in[19];
    float* logits = (float*)d_out;

    constexpr int B = 64, S = 197, SP = 224, D = 768, Hh = 12, DH = 64, L = 12,
                  MLP = 3072, NP = 196, OUTD = 1000;
    const int Mtok = B * S;     // 12608
    const int Mpat = B * NP;    // 12544

    char* wsp = (char*)d_ws;
    auto carve = [&](size_t bytes) {
        void* r = (void*)wsp;
        wsp += (bytes + 255) & ~(size_t)255;
        return r;
    };
    _Float16* APATCH = (_Float16*)carve((size_t)Mpat * D * 2);
    float*    OUT    = (float*)   carve((size_t)Mtok * D * 4);
    _Float16* XLN    = (_Float16*)carve((size_t)Mtok * D * 2);
    _Float16* Qb     = (_Float16*)carve((size_t)Mtok * D * 2);
    _Float16* Kb     = (_Float16*)carve((size_t)Mtok * D * 2);
    _Float16* Vt     = (_Float16*)carve((size_t)B * Hh * DH * SP * 2);  // transposed V
    float*    Sc     = (float*)   carve((size_t)B * Hh * S * SP * 4);
    _Float16* Pb     = (_Float16*)carve((size_t)B * Hh * S * SP * 2);
    _Float16* H1     = (_Float16*)carve((size_t)Mtok * MLP * 2);
    _Float16* ACLS   = (_Float16*)carve((size_t)B * D * 2);
    _Float16* WPh    = (_Float16*)carve((size_t)D * D * 2);             // N x K (native)
    _Float16* WQBD   = (_Float16*)carve((size_t)L * D * D * 2);
    _Float16* WKBD   = (_Float16*)carve((size_t)L * D * D * 2);
    _Float16* WVBD   = (_Float16*)carve((size_t)L * D * D * 2);
    _Float16* W1h    = (_Float16*)carve((size_t)L * MLP * D * 2);       // N x K (native)
    _Float16* W2h    = (_Float16*)carve((size_t)L * D * MLP * 2);       // N x K (native)
    _Float16* Whh    = (_Float16*)carve((size_t)OUTD * D * 2);          // N x K (native)
    float* TOK = Sc;  // patch-token temp aliases score buffer (free at that point)

    const dim3 blk(256);
    auto blocks1d = [](long n) { return dim3((unsigned)((n + 255) / 256)); };

    // ---- weight prep: fp32 -> f16, keeping native N x K layouts ----
    convert_f16_kernel<<<blocks1d((long)D * D), blk, 0, stream>>>(Wp, WPh, (long)D * D);
    convert_f16_kernel<<<blocks1d((long)L * MLP * D), blk, 0, stream>>>(W1, W1h, (long)L * MLP * D);
    convert_f16_kernel<<<blocks1d((long)L * D * MLP), blk, 0, stream>>>(W2, W2h, (long)L * D * MLP);
    convert_f16_kernel<<<blocks1d((long)OUTD * D), blk, 0, stream>>>(Wh, Whh, (long)OUTD * D);
    blockdiag_kernel<<<blocks1d((long)L * D * D), blk, 0, stream>>>(Wq, WQBD, (long)L * D * D);
    blockdiag_kernel<<<blocks1d((long)L * D * D), blk, 0, stream>>>(Wk, WKBD, (long)L * D * D);
    blockdiag_kernel<<<blocks1d((long)L * D * D), blk, 0, stream>>>(Wv, WVBD, (long)L * D * D);
    zero_vt_pad_kernel<<<blocks1d((long)B * Hh * DH * (SP - S)), blk, 0, stream>>>(
        Vt, (long)B * Hh * DH * (SP - S));

    // ---- patch embedding ----
    im2col_kernel<<<blocks1d((long)Mpat * D), blk, 0, stream>>>(x, APATCH, (long)Mpat * D);
    {
        dim3 g((D + 63) / 64, (Mpat + 127) / 128, 1);
        gemm_wmma<false, true, false, false, false><<<g, blk, 0, stream>>>(
            APATCH, WPh, TOK, nullptr, bp, nullptr,
            Mpat, D, D, D, D, D, 1.0f, 1, 0, 0, 0, 0, 0, 0);
    }
    assemble_tokens_kernel<<<blocks1d((long)Mtok * D), blk, 0, stream>>>(
        TOK, cls, OUT, (long)Mtok * D);

    // ---- transformer blocks ----
    const dim3 gD((D + 63) / 64, (Mtok + 127) / 128, 1);      // [Mtok x 768]
    const dim3 gM((MLP + 63) / 64, (Mtok + 127) / 128, 1);    // [Mtok x 3072]
    const dim3 gS((S + 63) / 64, (S + 127) / 128, B * Hh);    // scores
    const dim3 gV((DH + 63) / 64, (S + 127) / 128, B * Hh);   // att @ V
    for (int l = 0; l < L; ++l) {
        layernorm_kernel<<<Mtok, blk, 0, stream>>>(OUT, ln1g + l * D, ln1b + l * D, XLN);

        gemm_wmma<true, false, false, false, false><<<gD, blk, 0, stream>>>(
            XLN, WQBD + (size_t)l * D * D, nullptr, Qb, bq + l * D, nullptr,
            Mtok, D, D, D, D, D, 1.0f, 1, 0, 0, 0, 0, 0, 0);
        gemm_wmma<true, false, false, false, false><<<gD, blk, 0, stream>>>(
            XLN, WKBD + (size_t)l * D * D, nullptr, Kb, bk + l * D, nullptr,
            Mtok, D, D, D, D, D, 1.0f, 1, 0, 0, 0, 0, 0, 0);
        // V projection stored transposed+padded: Vt[n][h][o][224]
        gemm_wmma<true, false, false, false, true><<<gD, blk, 0, stream>>>(
            XLN, WVBD + (size_t)l * D * D, nullptr, Vt, bv + l * D, nullptr,
            Mtok, D, D, D, D, D, 1.0f, 1, 0, 0, 0, 0, 0, 0);

        // S[n,h,q,k] = (Q . K) / sqrt(64); K read directly as N x K
        gemm_wmma<false, true, false, false, false><<<gS, blk, 0, stream>>>(
            Qb, Kb, Sc, nullptr, nullptr, nullptr,
            S, S, DH, D, D, SP, 0.125f,
            Hh, (long)S * D, 64, (long)S * D, 64, (long)Hh * S * SP, (long)S * SP);

        softmax_kernel<<<B * Hh * S, blk, 0, stream>>>(Sc, Pb);

        // OUT += P @ V  (B = Vt, N x K layout, K padded to 224 with zeros both sides)
        gemm_wmma<false, true, false, true, false><<<gV, blk, 0, stream>>>(
            Pb, Vt, OUT, nullptr, nullptr, OUT,
            S, DH, SP, SP, SP, D, 1.0f,
            Hh, (long)Hh * S * SP, (long)S * SP, (long)Hh * DH * SP, (long)DH * SP,
            (long)S * D, 64);

        layernorm_kernel<<<Mtok, blk, 0, stream>>>(OUT, ln2g + l * D, ln2b + l * D, XLN);

        // fc1 + exact GELU -> f16
        gemm_wmma<true, false, true, false, false><<<gM, blk, 0, stream>>>(
            XLN, W1h + (size_t)l * MLP * D, nullptr, H1, b1 + l * MLP, nullptr,
            Mtok, MLP, D, D, D, MLP, 1.0f, 1, 0, 0, 0, 0, 0, 0);

        // fc2 + residual -> f32 OUT
        gemm_wmma<false, true, false, true, false><<<gD, blk, 0, stream>>>(
            H1, W2h + (size_t)l * D * MLP, OUT, nullptr, b2 + l * D, OUT,
            Mtok, D, MLP, MLP, MLP, D, 1.0f, 1, 0, 0, 0, 0, 0, 0);
    }

    // ---- classification head on CLS tokens ----
    extract_cls_kernel<<<blocks1d((long)B * D), blk, 0, stream>>>(OUT, ACLS, B * D);
    {
        dim3 g((OUTD + 63) / 64, (B + 127) / 128, 1);
        gemm_wmma<false, true, false, false, false><<<g, blk, 0, stream>>>(
            ACLS, Whh, logits, nullptr, bh, nullptr,
            B, OUTD, D, D, D, OUTD, 1.0f, 1, 0, 0, 0, 0, 0, 0);
    }
}
